// MinimalGRUCell_55722905699023
// MI455X (gfx1250) — matrix operand order repaired and checked by reading, not compiled, not run on hardware
//
#include <hip/hip_runtime.h>
#include <hip/hip_bf16.h>

// ---------------------------------------------------------------------------
// MinimalGRU forward for MI455X (gfx1250, wave32, WMMA).
//   B=8, L=4096, D=256, H=512, O=256, M=B*L=32768
// GEMMs in bf16 WMMA (v_wmma_f32_16x16x32_bf16, f32 accum) with
// GLOBAL_LOAD_ASYNC_TO_LDS_B128 (saddr form) double-buffered staging.
// Scan: 3-phase chunked log-space scan (serial chain 4096 -> 512 steps).
// ---------------------------------------------------------------------------

typedef __attribute__((ext_vector_type(16))) __bf16 v16bf;
typedef __attribute__((ext_vector_type(8)))  __bf16 v8bf;
typedef __attribute__((ext_vector_type(8)))  float  v8f;

#define BB 8
#define LL 4096
#define DD 256
#define HH 512
#define OO 256
#define MM (BB * LL)

#define SCH 16              // scan chunks per channel
#define SCT (LL / SCH)      // 256 time-steps per chunk

// ---- f32 -> bf16 (round-to-nearest-even), pure integer ops -----------------
__device__ __forceinline__ __bf16 to_bf16(float f) {
  union { float f; unsigned int u; } v; v.f = f;
  unsigned int u = v.u;
  u += 0x7FFFu + ((u >> 16) & 1u);
  unsigned short s = (unsigned short)(u >> 16);
  union { unsigned short s; __bf16 b; } r; r.s = s;
  return r.b;
}

__global__ __launch_bounds__(256) void cvt_f32_bf16(const float* __restrict__ in,
                                                    __bf16* __restrict__ out,
                                                    size_t n) {
  size_t i = (size_t)blockIdx.x * blockDim.x + threadIdx.x;
  size_t stride = (size_t)gridDim.x * blockDim.x;
  for (; i < n; i += stride) out[i] = to_bf16(in[i]);
}

// ---- CDNA5 async global->LDS, GVS form: mem = SGPR64 + VGPR32 --------------
// LDS destination address is taken from the vdst VGPR; flat shared ptr low 32 bits == LDS byte
// offset (ISA §10.2: LDS aperture truncates flat addr to addr[31:0]).
__device__ __forceinline__ void async_ld_b128(unsigned lds_off,
                                              const void* sbase,
                                              unsigned voff_bytes) {
  asm volatile("global_load_async_to_lds_b128 %0, %1, %2"
               :: "v"(lds_off), "v"(voff_bytes), "s"(sbase)
               : "memory");
}
__device__ __forceinline__ void wait_async0() {
  asm volatile("s_wait_asynccnt 0x0" ::: "memory");
}

// ---------------------------------------------------------------------------
// Tiled bf16 WMMA GEMM:  C[M,N] = act( A[M,K] @ W[N,K]^T + bias[N] )
//   block: 256 threads = 8 waves, macro-tile 64(M) x 128(N)
//   wave w: rows (w&1)*32, cols (w>>1)*32 -> 2x2 WMMA tiles, 4 accumulators
//   K step 32, double-buffered LDS via async-to-LDS, rows padded to 40 bf16
//   (80 B -> stride 20 dwords, conflict-free b128 fragment loads)
// ---------------------------------------------------------------------------
#define LDSTR 40

template <int K, bool RELU, bool OUT_BF16>
__global__ __launch_bounds__(256) void gemm_bf16_wmma(
    const __bf16* __restrict__ A,      // M x K, row-major
    const __bf16* __restrict__ W,      // N x K, row-major (acts as B^T)
    const float*  __restrict__ bias,   // N
    __bf16*       __restrict__ outB,   // M x N (if OUT_BF16)
    float*        __restrict__ outF,   // M x N (if !OUT_BF16)
    int N) {
  __shared__ __bf16 shA[2][64 * LDSTR];    // A tile: 64 rows x 32 K
  __shared__ __bf16 shB[2][128 * LDSTR];   // B tile: 128 rows(N) x 32 K

  const int tid  = threadIdx.x;
  const int lane = tid & 31;
  const int w    = tid >> 5;
  const int m0   = blockIdx.y * 64;
  const int n0   = blockIdx.x * 128;

  // staging split: A = 64 rows * 4 chunks (1/thread), B = 128 rows * 4 chunks
  // (2/thread); all chunks 16B
  const int arow = tid >> 2;            // 0..63
  const int acol = (tid & 3) * 8;
  const int brow = tid >> 1;            // 0..127
  const int bcol = (tid & 1) * 16;

  const unsigned aoff0 = (unsigned)(((m0 + arow) * K + acol) * 2);  // bytes
  const unsigned boff0 = (unsigned)(((n0 + brow) * K + bcol) * 2);
  const unsigned ldsA  = (unsigned)(unsigned long long)&shA[0][arow * LDSTR + acol];
  const unsigned ldsB  = (unsigned)(unsigned long long)&shB[0][brow * LDSTR + bcol];
  const unsigned bufAstride = (unsigned)(64 * LDSTR * 2);
  const unsigned bufBstride = (unsigned)(128 * LDSTR * 2);

  const __bf16* Agp = A + (size_t)(m0 + arow) * K + acol;   // for prefetch only
  const __bf16* Wgp = W + (size_t)(n0 + brow) * K + bcol;

  auto stage = [&](int buf, int kstep) {
    const unsigned kb = (unsigned)(kstep * 64);             // 32 elems * 2 B
    async_ld_b128(ldsA + buf * bufAstride, A, aoff0 + kb);
    async_ld_b128(ldsB + buf * bufBstride, W, boff0 + kb);
    async_ld_b128(ldsB + buf * bufBstride + 16, W, boff0 + kb + 16);
  };

  stage(0, 0);
  wait_async0();
  __syncthreads();

  const int mt   = (w & 1) * 32;
  const int nt   = (w >> 1) * 32;
  const int half = lane >> 4;     // 0: lanes 0-15, 1: lanes 16-31
  const int l16  = lane & 15;

  v8f acc[2][2] = {};

  constexpr int NK = K / 32;
#pragma unroll
  for (int i = 0; i < NK; ++i) {
    const int cur = i & 1;
    if (i + 1 < NK) stage(cur ^ 1, i + 1);          // async, overlaps compute
    if (i + 2 < NK) {
      __builtin_prefetch(Agp + (i + 2) * 32, 0, 1); // global_prefetch_b8 -> L2
      __builtin_prefetch(Wgp + (i + 2) * 32, 0, 1);
    }

    const __bf16* As_ = &shA[cur][0];
    const __bf16* Bs_ = &shB[cur][0];

    // A fragments (16-bit 16x32 layout, ISA 7.12.2):
    //   lanes 0-15: elems 0-7 -> K 0..7,  8-15 -> K 16..23
    //   lanes16-31: elems 0-7 -> K 8..15, 8-15 -> K 24..31
    v16bf af[2];
#pragma unroll
    for (int p = 0; p < 2; ++p) {
      const int off = (mt + p * 16 + l16) * LDSTR + half * 8;
      v8bf lo = *(const v8bf*)(As_ + off);
      v8bf hi = *(const v8bf*)(As_ + off + 16);
#pragma unroll
      for (int e = 0; e < 8; ++e) { af[p][e] = lo[e]; af[p][e + 8] = hi[e]; }
    }
    // B fragments (32x16): lane = column, elems 0-7 -> K half*8.., 8-15 -> K 16+half*8..
    v16bf bfr[2];
#pragma unroll
    for (int q = 0; q < 2; ++q) {
      const int off = (nt + q * 16 + l16) * LDSTR + half * 8;
      v8bf lo = *(const v8bf*)(Bs_ + off);
      v8bf hi = *(const v8bf*)(Bs_ + off + 16);
#pragma unroll
      for (int e = 0; e < 8; ++e) { bfr[q][e] = lo[e]; bfr[q][e + 8] = hi[e]; }
    }

#pragma unroll
    for (int p = 0; p < 2; ++p)
#pragma unroll
      for (int q = 0; q < 2; ++q)
        acc[p][q] = __builtin_amdgcn_wmma_f32_16x16x32_bf16(
            false, af[p], false, bfr[q], (short)0, acc[p][q], false, false);

    if (i + 1 < NK) {
      wait_async0();      // this wave's async tile writes have landed in LDS
      __syncthreads();    // ... and everyone else's
    }
  }

  // Epilogue: C/D f32 16x16 layout — VGPR r: lanes0-15 -> M=r, lanes16-31 -> M=8+r
#pragma unroll
  for (int p = 0; p < 2; ++p) {
#pragma unroll
    for (int q = 0; q < 2; ++q) {
      const int gn = n0 + nt + q * 16 + l16;
      const float bv = bias[gn];
#pragma unroll
      for (int r = 0; r < 8; ++r) {
        const int gm = m0 + mt + p * 16 + half * 8 + r;
        float v = acc[p][q][r] + bv;
        if (RELU) v = v > 0.0f ? v : 0.0f;
        if (OUT_BF16) outB[(size_t)gm * N + gn] = to_bf16(v);
        else          outF[(size_t)gm * N + gn] = v;
      }
    }
  }
}

// ---------------------------------------------------------------------------
// 3-phase chunked log-space minGRU scan.
// Recurrence (identical to single-pass reference ordering, checkpointed):
//   A += logsig(-z);  x = (logsig(z) + g(ht)) - A
//   (m, s) running-max logsumexp of x;  h = exp(A + m) * s
// Phase 1: per-chunk local scan from (A=0, m=-inf, s=0)  -> (Asum, m, s)
// Phase 2: per-channel 16-step exclusive merge            -> (preA, preM, preS)
// Phase 3: replay each chunk seeded with carried state, write h
// latent: [B, L, 2H] f32; h_out: [B, L, H] f32; chunk state: [B, SCH, H]
// ---------------------------------------------------------------------------
__device__ __forceinline__ float logsigmoidf(float v) {
  return (v <= 0.0f) ? (v - log1pf(__expf(v))) : (-log1pf(__expf(-v)));
}

__device__ __forceinline__ void scan_step(const float* __restrict__ row,
                                          float& A, float& m, float& s) {
  const float z  = row[0];
  const float ht = row[HH];
  const float zlog = logsigmoidf(z);    // -softplus(-z)
  const float zinv = logsigmoidf(-z);   // -softplus(z)
  const float g    = (ht >= 0.0f) ? __logf(ht + 0.5f) : logsigmoidf(ht);
  A += zinv;
  const float x = (zlog + g) - A;
  if (x <= m) {
    s += __expf(x - m);
  } else {
    s = s * __expf(m - x) + 1.0f;
    m = x;
  }
}

__global__ __launch_bounds__(256) void scan_local(const float* __restrict__ latent,
                                                  float* __restrict__ Asum,
                                                  float* __restrict__ Mloc,
                                                  float* __restrict__ Sloc) {
  const int c  = blockIdx.z;
  const int b  = blockIdx.y;
  const int ch = blockIdx.x * blockDim.x + threadIdx.x;

  float A = 0.0f, m = -3.0e38f, s = 0.0f;
  const float* lz = latent + ((size_t)b * LL + (size_t)c * SCT) * (2 * HH) + ch;
  for (int t = 0; t < SCT; ++t) scan_step(lz + (size_t)t * (2 * HH), A, m, s);

  const int idx = (b * SCH + c) * HH + ch;
  Asum[idx] = A; Mloc[idx] = m; Sloc[idx] = s;
}

__global__ __launch_bounds__(256) void scan_prefix(const float* __restrict__ h0,
                                                   const float* __restrict__ Asum,
                                                   const float* __restrict__ Mloc,
                                                   const float* __restrict__ Sloc,
                                                   float* __restrict__ preA,
                                                   float* __restrict__ preM,
                                                   float* __restrict__ preS) {
  const int g  = blockIdx.x * blockDim.x + threadIdx.x;   // 0 .. B*H-1
  const int b  = g / HH;
  const int ch = g % HH;

  float A = 0.0f;
  float m = h0[b * HH + ch];   // lv[0] = h_0 (log-space), A[0] = 0
  float s = 1.0f;
  for (int c = 0; c < SCH; ++c) {
    const int idx = (b * SCH + c) * HH + ch;
    preA[idx] = A; preM[idx] = m; preS[idx] = s;
    const float mc = Mloc[idx] - A;      // shift chunk-local x by carried A
    const float sc = Sloc[idx];
    if (mc <= m) {
      s += sc * __expf(mc - m);
    } else {
      s = s * __expf(m - mc) + sc;
      m = mc;
    }
    A += Asum[idx];
  }
}

__global__ __launch_bounds__(256) void scan_apply(const float* __restrict__ latent,
                                                  const float* __restrict__ preA,
                                                  const float* __restrict__ preM,
                                                  const float* __restrict__ preS,
                                                  float* __restrict__ h_out) {
  const int c  = blockIdx.z;
  const int b  = blockIdx.y;
  const int ch = blockIdx.x * blockDim.x + threadIdx.x;

  const int idx = (b * SCH + c) * HH + ch;
  float A = preA[idx], m = preM[idx], s = preS[idx];

  const float* lz = latent + ((size_t)b * LL + (size_t)c * SCT) * (2 * HH) + ch;
  float* ho = h_out + ((size_t)b * LL + (size_t)c * SCT) * HH + ch;
  for (int t = 0; t < SCT; ++t) {
    scan_step(lz + (size_t)t * (2 * HH), A, m, s);
    ho[(size_t)t * HH] = __expf(A + m) * s;
  }
}

// ---------------------------------------------------------------------------
// Tiny output head (f32 scalar; ~3 MFLOP total, irrelevant for runtime)
// ---------------------------------------------------------------------------
__global__ __launch_bounds__(256) void head1_kernel(const float* __restrict__ h_out,
                                                    const float* __restrict__ Wo1,
                                                    const float* __restrict__ bo1,
                                                    float* __restrict__ obuf) {
  __shared__ float hl[HH];
  const int b = blockIdx.x;
  const float* hlast = h_out + ((size_t)b * LL + (LL - 1)) * HH;
  for (int k = threadIdx.x; k < HH; k += blockDim.x) hl[k] = hlast[k];
  __syncthreads();
  for (int n = threadIdx.x; n < HH; n += blockDim.x) {
    const float* wr = Wo1 + (size_t)n * HH;
    float acc = bo1[n];
    for (int k = 0; k < HH; ++k) acc += hl[k] * wr[k];
    obuf[b * HH + n] = acc > 0.0f ? acc : 0.0f;
  }
}

__global__ __launch_bounds__(256) void head2_kernel(const float* __restrict__ obuf,
                                                    const float* __restrict__ Wo2,
                                                    const float* __restrict__ bo2,
                                                    float* __restrict__ out) {
  __shared__ float ob[HH];
  const int b = blockIdx.x;
  for (int k = threadIdx.x; k < HH; k += blockDim.x) ob[k] = obuf[b * HH + k];
  __syncthreads();
  for (int j = threadIdx.x; j < OO; j += blockDim.x) {
    const float* wr = Wo2 + (size_t)j * HH;
    float acc = bo2[j];
    for (int k = 0; k < HH; ++k) acc += ob[k] * wr[k];
    out[b * OO + j] = acc;
  }
}

// ---------------------------------------------------------------------------
extern "C" void kernel_launch(void* const* d_in, const int* in_sizes, int n_in,
                              void* d_out, int out_size, void* d_ws, size_t ws_size,
                              hipStream_t stream) {
  const float* x   = (const float*)d_in[0];
  const float* h0  = (const float*)d_in[1];
  const float* W1  = (const float*)d_in[2];
  const float* b1  = (const float*)d_in[3];
  const float* W2  = (const float*)d_in[4];
  const float* b2  = (const float*)d_in[5];
  const float* W3  = (const float*)d_in[6];
  const float* b3  = (const float*)d_in[7];
  const float* Wo1 = (const float*)d_in[8];
  const float* bo1 = (const float*)d_in[9];
  const float* Wo2 = (const float*)d_in[10];
  const float* bo2 = (const float*)d_in[11];

  float* out   = (float*)d_out;            // [B, O]
  float* h_ret = (float*)d_out + BB * OO;  // [B, L, H]

  // workspace carve-out (256B aligned regions)
  char* ws = (char*)d_ws;
  size_t off = 0;
  auto carve = [&](size_t bytes) -> void* {
    void* p = ws + off;
    off = (off + bytes + 255) & ~(size_t)255;
    return p;
  };
  __bf16* xb   = (__bf16*)carve((size_t)MM * DD * 2);
  __bf16* w1b  = (__bf16*)carve((size_t)HH * DD * 2);
  __bf16* w2b  = (__bf16*)carve((size_t)HH * HH * 2);
  __bf16* w3b  = (__bf16*)carve((size_t)2 * HH * HH * 2);
  __bf16* t1   = (__bf16*)carve((size_t)MM * HH * 2);
  __bf16* t2   = (__bf16*)carve((size_t)MM * HH * 2);
  float*  lat  = (float*) carve((size_t)MM * 2 * HH * 4);
  float*  obuf = (float*) carve((size_t)BB * HH * 4);
  const size_t cs = (size_t)BB * SCH * HH;      // chunk-state elements
  float* Asum = (float*)carve(cs * 4);
  float* Mloc = (float*)carve(cs * 4);
  float* Sloc = (float*)carve(cs * 4);
  float* preA = (float*)carve(cs * 4);
  float* preM = (float*)carve(cs * 4);
  float* preS = (float*)carve(cs * 4);

  // 1) precision conversion
  cvt_f32_bf16<<<2048, 256, 0, stream>>>(x,  xb,  (size_t)MM * DD);
  cvt_f32_bf16<<<128,  256, 0, stream>>>(W1, w1b, (size_t)HH * DD);
  cvt_f32_bf16<<<256,  256, 0, stream>>>(W2, w2b, (size_t)HH * HH);
  cvt_f32_bf16<<<512,  256, 0, stream>>>(W3, w3b, (size_t)2 * HH * HH);

  // 2) GEMM chain (WMMA bf16, f32 accumulate), macro-tile 64x128
  gemm_bf16_wmma<DD, true,  true ><<<dim3(HH / 128,     MM / 64), 256, 0, stream>>>(
      xb, w1b, b1, t1, nullptr, HH);
  gemm_bf16_wmma<HH, true,  true ><<<dim3(HH / 128,     MM / 64), 256, 0, stream>>>(
      t1, w2b, b2, t2, nullptr, HH);
  gemm_bf16_wmma<HH, false, false><<<dim3(2 * HH / 128, MM / 64), 256, 0, stream>>>(
      t2, w3b, b3, nullptr, lat, 2 * HH);

  // 3) chunked log-space scan -> h (written straight into d_out)
  scan_local <<<dim3(HH / 256, BB, SCH), 256, 0, stream>>>(lat, Asum, Mloc, Sloc);
  scan_prefix<<<(BB * HH) / 256, 256, 0, stream>>>(h0, Asum, Mloc, Sloc,
                                                   preA, preM, preS);
  scan_apply <<<dim3(HH / 256, BB, SCH), 256, 0, stream>>>(lat, preA, preM, preS,
                                                           h_ret);

  // 4) output head
  head1_kernel<<<BB, 256, 0, stream>>>(h_ret, Wo1, bo1, obuf);
  head2_kernel<<<BB, 256, 0, stream>>>(obuf, Wo2, bo2, out);
}
